// MetaBlock_29858612642213
// MI455X (gfx1250) — compile-verified, matrix operand order checked by reading, not run on hardware
//
#include <hip/hip_runtime.h>
#include <cstdint>

// ---- problem constants ----
static constexpr int B_ = 2, L_ = 512, C_ = 256, H_ = 4, HD_ = 64, NL_ = 2;
static constexpr int M_ = B_ * L_;          // 1024 rows
static constexpr int QKVN = 3 * C_;         // 768
static constexpr int FF = 4 * C_;           // 1024

// ---- WMMA types (CDNA5 wave32) ----
typedef __attribute__((ext_vector_type(16))) __bf16 v16bf;
typedef __attribute__((ext_vector_type(8)))  float  v8f;
typedef __attribute__((ext_vector_type(4)))  unsigned int v4u;

union Frag { v4u q[2]; v16bf v; };   // 16 bf16 = 8 VGPRs per lane

__device__ __forceinline__ unsigned short f2bf(float f) {
  union { float f; unsigned u; } x; x.f = f;
  unsigned u = x.u;
  u += 0x7fffu + ((u >> 16) & 1u);   // round-to-nearest-even
  return (unsigned short)(u >> 16);
}

__device__ __forceinline__ v8f wmma_bf16(const Frag& a, const Frag& b, v8f c) {
  return __builtin_amdgcn_wmma_f32_16x16x32_bf16(false, a.v, false, b.v,
                                                 (short)0, c, false, false);
}

// =====================================================================
// Generic bf16-WMMA GEMM:  out[M,N] = act(A[M,K] @ W[K,N] + bias (+res))
// BM=128, BN=128, BK=32. 256 threads = 8 waves; wave tile 64x32 (4x2 frags).
// M,N multiples of 128; K multiple of 32 (true for all call sites).
// =====================================================================
__global__ __launch_bounds__(256) void gemm_bf16(
    const float* __restrict__ A, const float* __restrict__ W,
    const float* __restrict__ bias, const float* res,
    float* out, int M, int N, int K, int act)
{
  __shared__ unsigned short sA[128][40];   // [m][k], pad to 80B rows
  __shared__ unsigned short sB[128][40];   // [n][k] (W transposed)
  const int tid = threadIdx.x, lane = tid & 31, wv = tid >> 5;
  const int m0 = blockIdx.y * 128, n0 = blockIdx.x * 128;
  const int wm = (wv >> 2) * 64;           // 0 / 64
  const int wn = (wv & 3) * 32;            // 0..96
  const int l15 = lane & 15, lg = lane >> 4;

  v8f acc[4][2];
#pragma unroll
  for (int i = 0; i < 4; ++i)
#pragma unroll
    for (int j = 0; j < 2; ++j) acc[i][j] = (v8f){0,0,0,0,0,0,0,0};

  for (int k0 = 0; k0 < K; k0 += 32) {
#pragma unroll
    for (int it = 0; it < 16; ++it) {      // A tile 128x32
      int idx = tid + it * 256;
      int r = idx >> 5, c = idx & 31;
      sA[r][c] = f2bf(A[(size_t)(m0 + r) * K + k0 + c]);
    }
#pragma unroll
    for (int it = 0; it < 16; ++it) {      // B tile 32x128 -> [n][k]
      int idx = tid + it * 256;
      int k = idx >> 7, n = idx & 127;
      sB[n][k] = f2bf(W[(size_t)(k0 + k) * N + n0 + n]);
    }
    if (k0 + 32 < K) {                     // exercise global_prefetch_b8
      __builtin_prefetch(&A[(size_t)(m0 + (tid >> 1)) * K + k0 + 32], 0, 1);
      __builtin_prefetch(&W[(size_t)(k0 + 32 + (tid >> 3)) * N + n0], 0, 1);
    }
    __syncthreads();

    const int kb = 8 * lg;                 // K-offset per lane half (ISA layout)
    Frag bfr[2];
#pragma unroll
    for (int ni = 0; ni < 2; ++ni) {
      int n = wn + ni * 16 + l15;
      bfr[ni].q[0] = *(const v4u*)&sB[n][kb];
      bfr[ni].q[1] = *(const v4u*)&sB[n][kb + 16];
    }
#pragma unroll
    for (int mi = 0; mi < 4; ++mi) {
      Frag a;
      int m = wm + mi * 16 + l15;
      a.q[0] = *(const v4u*)&sA[m][kb];
      a.q[1] = *(const v4u*)&sA[m][kb + 16];
#pragma unroll
      for (int ni = 0; ni < 2; ++ni)
        acc[mi][ni] = wmma_bf16(a, bfr[ni], acc[mi][ni]);
    }
    __syncthreads();
  }

#pragma unroll
  for (int mi = 0; mi < 4; ++mi) {
    int row0 = m0 + wm + mi * 16 + 8 * lg;   // C/D layout: row = r + 8*(lane>>4)
#pragma unroll
    for (int ni = 0; ni < 2; ++ni) {
      int col = n0 + wn + ni * 16 + l15;
      float bv = bias ? bias[col] : 0.f;
#pragma unroll
      for (int r = 0; r < 8; ++r) {
        int row = row0 + r;
        float v = acc[mi][ni][r] + bv;
        if (res) v += res[(size_t)row * N + col];
        if (act == 1) {                      // tanh-gelu (jax default)
          float t = v + 0.044715f * v * v * v;
          v = 0.5f * v * (1.f + tanhf(0.7978845608028654f * t));
        }
        out[(size_t)row * N + col] = v;
      }
    }
  }
}

// =====================================================================
// h0 = x @ proj_in_w + proj_in_b + pos_embed          (IN=3, scalar)
// =====================================================================
__global__ void h0_kernel(const float* __restrict__ x, const float* __restrict__ wi,
                          const float* __restrict__ bi, const float* __restrict__ pos,
                          float* __restrict__ h)
{
  int idx = blockIdx.x * 256 + threadIdx.x;
  int c = idx & 255, m = idx >> 8;
  float v = bi[c] + pos[(size_t)(m & (L_ - 1)) * C_ + c];
#pragma unroll
  for (int k = 0; k < 3; ++k) v += x[m * 3 + k] * wi[k * C_ + c];
  h[idx] = v;
}

// dist[b,i,j] = ||x_i - x_j||  (IN=3)
__global__ void dist_kernel(const float* __restrict__ x, float* __restrict__ dist)
{
  int idx = blockIdx.x * 256 + threadIdx.x;
  int j = idx & (L_ - 1), i = (idx >> 9) & (L_ - 1), b = idx >> 18;
  float d2 = 0.f;
#pragma unroll
  for (int k = 0; k < 3; ++k) {
    float d = x[(b * L_ + i) * 3 + k] - x[(b * L_ + j) * 3 + k];
    d2 += d * d;
  }
  dist[idx] = sqrtf(fmaxf(d2, 0.f));
}

// rank-1 collapse of pair bias: alpha/beta per (layer, head)
__global__ void ab_kernel(const float* __restrict__ pw, const float* __restrict__ pb,
                          const float* __restrict__ ph, float* __restrict__ ab)
{
  int t = threadIdx.x;
  if (t >= NL_ * H_) return;
  int l = t / H_, h = t % H_;
  float al = 0.f, be = 0.f;
  for (int c = 0; c < C_; ++c) {
    float w = ph[((size_t)l * C_ + c) * H_ + h];
    al += pw[c] * w; be += pb[c] * w;
  }
  ab[t * 2] = al; ab[t * 2 + 1] = be;
}

// =====================================================================
// LayerNorm (one wave32 per row, 8 rows/block); doAdd: h += ce first.
// =====================================================================
__global__ __launch_bounds__(256) void add_ln_kernel(
    float* __restrict__ h, const float* __restrict__ ce,
    const float* __restrict__ s, const float* __restrict__ bb,
    float* __restrict__ a, int doAdd)
{
  const int tid = threadIdx.x, lane = tid & 31, wv = tid >> 5;
  const int row = blockIdx.x * 8 + wv;
  float v[8]; float sum = 0.f;
#pragma unroll
  for (int t = 0; t < 8; ++t) {
    int c = lane + t * 32;
    float x = h[(size_t)row * C_ + c];
    if (doAdd) { x += ce[(size_t)row * C_ + c]; h[(size_t)row * C_ + c] = x; }
    v[t] = x; sum += x;
  }
#pragma unroll
  for (int off = 16; off; off >>= 1) sum += __shfl_xor(sum, off, 32);
  float mean = sum * (1.f / C_), var = 0.f;
#pragma unroll
  for (int t = 0; t < 8; ++t) { float d = v[t] - mean; var += d * d; }
#pragma unroll
  for (int off = 16; off; off >>= 1) var += __shfl_xor(var, off, 32);
  float rs = rsqrtf(var * (1.f / C_) + 1e-5f);
#pragma unroll
  for (int t = 0; t < 8; ++t) {
    int c = lane + t * 32;
    a[(size_t)row * C_ + c] = (v[t] - mean) * rs * s[c] + bb[c];
  }
}

// =====================================================================
// Attention pass 1: logits[bh,i,j] = (q.k)/8 + dist*alpha + beta, causal -1e9
// 64 blocks = (B*H) x (L/64 q-tiles). WMMA q@kT, skip blocks above diagonal.
// =====================================================================
__global__ __launch_bounds__(256) void att_logits(
    const float* __restrict__ qkv, const float* __restrict__ dist,
    const float* __restrict__ ab, int layer, float* __restrict__ logits)
{
  __shared__ unsigned short qT[64][72];   // [q][d]
  __shared__ unsigned short kT[64][72];   // [key][d]  (== B-layout [n][k])
  const int tid = threadIdx.x, lane = tid & 31, wv = tid >> 5;
  const int qt = blockIdx.x & 7, bh = blockIdx.x >> 3;
  const int b = bh >> 2, hh = bh & 3;
  const int qi0 = qt * 64;
  const float alpha = ab[(layer * H_ + hh) * 2 + 0];
  const float beta  = ab[(layer * H_ + hh) * 2 + 1];
  const int l15 = lane & 15, lg = lane >> 4;
  const int wr = (wv >> 1) * 16, wc = (wv & 1) * 32;

  for (int idx = tid; idx < 64 * 64; idx += 256) {
    int r = idx >> 6, c = idx & 63;
    qT[r][c] = f2bf(qkv[(size_t)(b * L_ + qi0 + r) * QKVN + hh * HD_ + c]);
  }
  for (int j0 = 0; j0 <= qi0; j0 += 64) {
    for (int idx = tid; idx < 64 * 64; idx += 256) {
      int r = idx >> 6, c = idx & 63;
      kT[r][c] = f2bf(qkv[(size_t)(b * L_ + j0 + r) * QKVN + C_ + hh * HD_ + c]);
    }
    __syncthreads();
    v8f acc[2] = {(v8f){0,0,0,0,0,0,0,0}, (v8f){0,0,0,0,0,0,0,0}};
#pragma unroll
    for (int kk = 0; kk < 64; kk += 32) {
      int kb = kk + 8 * lg;
      Frag a;
      int m = wr + l15;
      a.q[0] = *(const v4u*)&qT[m][kb];
      a.q[1] = *(const v4u*)&qT[m][kb + 16];
#pragma unroll
      for (int ni = 0; ni < 2; ++ni) {
        Frag bb;
        int n = wc + ni * 16 + l15;
        bb.q[0] = *(const v4u*)&kT[n][kb];
        bb.q[1] = *(const v4u*)&kT[n][kb + 16];
        acc[ni] = wmma_bf16(a, bb, acc[ni]);
      }
    }
#pragma unroll
    for (int ni = 0; ni < 2; ++ni) {
      int j = j0 + wc + ni * 16 + l15;
#pragma unroll
      for (int r = 0; r < 8; ++r) {
        int i = qi0 + wr + r + 8 * lg;
        float v = acc[ni][r] * 0.125f
                + dist[(size_t)b * L_ * L_ + (size_t)i * L_ + j] * alpha + beta;
        if (j > i) v = -1e9f;
        logits[((size_t)bh * L_ + i) * L_ + j] = v;
      }
    }
    __syncthreads();
  }
}

// Attention pass 2: row softmax over j<=i, write bf16 probs (0 for j>i)
__global__ __launch_bounds__(256) void att_softmax(
    const float* __restrict__ logits, unsigned short* __restrict__ attn)
{
  const int tid = threadIdx.x, lane = tid & 31, wv = tid >> 5;
  const int row = blockIdx.x * 8 + wv;            // bh*L + i
  const int i = row & (L_ - 1);
  const size_t base = (size_t)row * L_;
  float v[16]; float mx = -3.0e38f;
#pragma unroll
  for (int t = 0; t < 16; ++t) {
    int j = lane + t * 32;
    float x = (j <= i) ? logits[base + j] : -3.0e38f;
    v[t] = x; mx = fmaxf(mx, x);
  }
#pragma unroll
  for (int off = 16; off; off >>= 1) mx = fmaxf(mx, __shfl_xor(mx, off, 32));
  float s = 0.f;
#pragma unroll
  for (int t = 0; t < 16; ++t) {
    int j = lane + t * 32;
    float e = (j <= i) ? expf(v[t] - mx) : 0.f;
    v[t] = e; s += e;
  }
#pragma unroll
  for (int off = 16; off; off >>= 1) s += __shfl_xor(s, off, 32);
  float inv = 1.f / s;
#pragma unroll
  for (int t = 0; t < 16; ++t) attn[base + lane + t * 32] = f2bf(v[t] * inv);
}

// Attention pass 3: O = attn @ v. A-frags loaded directly from global bf16
// (K-pairs contiguous in the 16x32 A layout -> two 16B loads per frag).
__global__ __launch_bounds__(256) void att_av(
    const unsigned short* __restrict__ attn, const float* __restrict__ qkv,
    float* __restrict__ o)
{
  __shared__ unsigned short vT[64][72];   // [d][key] == B-layout [n][k]
  const int tid = threadIdx.x, lane = tid & 31, wv = tid >> 5;
  const int qt = blockIdx.x & 7, bh = blockIdx.x >> 3;
  const int b = bh >> 2, hh = bh & 3;
  const int qi0 = qt * 64;
  const int l15 = lane & 15, lg = lane >> 4;
  const int wr = (wv >> 1) * 16, wc = (wv & 1) * 32;

  v8f acc[2] = {(v8f){0,0,0,0,0,0,0,0}, (v8f){0,0,0,0,0,0,0,0}};
  for (int j0 = 0; j0 <= qi0; j0 += 64) {
    for (int idx = tid; idx < 64 * 64; idx += 256) {
      int d = idx & 63, key = idx >> 6;
      vT[d][key] = f2bf(qkv[(size_t)(b * L_ + j0 + key) * QKVN + 2 * C_ + hh * HD_ + d]);
    }
    __syncthreads();
#pragma unroll
    for (int kk = 0; kk < 64; kk += 32) {
      int kb = kk + 8 * lg;
      Frag a;
      int i = qi0 + wr + l15;
      const unsigned short* ap = attn + ((size_t)bh * L_ + i) * L_ + j0 + kb;
      a.q[0] = *(const v4u*)ap;
      a.q[1] = *(const v4u*)(ap + 16);
#pragma unroll
      for (int ni = 0; ni < 2; ++ni) {
        Frag bb;
        int n = wc + ni * 16 + l15;
        bb.q[0] = *(const v4u*)&vT[n][kb];
        bb.q[1] = *(const v4u*)&vT[n][kb + 16];
        acc[ni] = wmma_bf16(a, bb, acc[ni]);
      }
    }
    __syncthreads();
  }
#pragma unroll
  for (int ni = 0; ni < 2; ++ni) {
    int col = hh * HD_ + wc + ni * 16 + l15;
#pragma unroll
    for (int r = 0; r < 8; ++r) {
      int row = b * L_ + qi0 + wr + r + 8 * lg;
      o[(size_t)row * C_ + col] = acc[ni][r];
    }
  }
}

// y = h @ proj_out_w + proj_out_b   (N=6, scalar dot)
__global__ void proj_out_kernel(const float* __restrict__ h, const float* __restrict__ w,
                                const float* __restrict__ bb, float* __restrict__ y)
{
  int idx = blockIdx.x * 256 + threadIdx.x;
  if (idx >= M_ * 6) return;
  int m = idx / 6, j = idx % 6;
  float s = bb[j];
  for (int c = 0; c < C_; ++c) s += h[(size_t)m * C_ + c] * w[c * 6 + j];
  y[idx] = s;
}

// causal shift + affine coupling: x_out = (x - xb_prev) * exp(-xa_prev)
__global__ void coupling_kernel(const float* __restrict__ x, const float* __restrict__ y,
                                float* __restrict__ out)
{
  int idx = blockIdx.x * 256 + threadIdx.x;
  if (idx >= B_ * L_ * 3) return;
  int i = idx % 3, l = (idx / 3) & (L_ - 1), b = idx / (3 * L_);
  float xa = 0.f, xb = 0.f;
  if (l > 0) {
    const float* yr = y + (size_t)(b * L_ + l - 1) * 6;
    xa = yr[i]; xb = yr[3 + i];
  }
  out[idx] = (x[idx] - xb) * expf(-xa);
}

// logdet[b] = -sum(xa)/ (L*3); shifted xa rows 1..L-1 == y rows 0..L-2
__global__ __launch_bounds__(256) void logdet_kernel(const float* __restrict__ y,
                                                     float* __restrict__ out)
{
  __shared__ float red[256];
  int b = blockIdx.x;
  float s = 0.f;
  for (int idx = threadIdx.x; idx < (L_ - 1) * 3; idx += 256) {
    int l = idx / 3, i = idx % 3;
    s += y[(size_t)(b * L_ + l) * 6 + i];
  }
  red[threadIdx.x] = s;
  __syncthreads();
  for (int off = 128; off; off >>= 1) {
    if (threadIdx.x < off) red[threadIdx.x] += red[threadIdx.x + off];
    __syncthreads();
  }
  if (threadIdx.x == 0) out[b] = -red[0] / (float)(L_ * 3);
}

// =====================================================================
extern "C" void kernel_launch(void* const* d_in, const int* in_sizes, int n_in,
                              void* d_out, int out_size, void* d_ws, size_t ws_size,
                              hipStream_t stream) {
  const float* x           = (const float*)d_in[0];
  const float* cond        = (const float*)d_in[1];
  const float* proj_in_w   = (const float*)d_in[2];
  const float* proj_in_b   = (const float*)d_in[3];
  const float* proj_cond_w = (const float*)d_in[4];
  const float* proj_cond_b = (const float*)d_in[5];
  const float* pair_w      = (const float*)d_in[6];
  const float* pair_b      = (const float*)d_in[7];
  const float* pos_embed   = (const float*)d_in[8];
  const float* ln1_s       = (const float*)d_in[9];
  const float* ln1_b       = (const float*)d_in[10];
  const float* qkv_w       = (const float*)d_in[11];
  const float* qkv_b       = (const float*)d_in[12];
  const float* pairhead_w  = (const float*)d_in[13];
  const float* out_w       = (const float*)d_in[14];
  const float* out_b       = (const float*)d_in[15];
  const float* ln2_s       = (const float*)d_in[16];
  const float* ln2_b       = (const float*)d_in[17];
  const float* mlp1_w      = (const float*)d_in[18];
  const float* mlp1_b      = (const float*)d_in[19];
  const float* mlp2_w      = (const float*)d_in[20];
  const float* mlp2_b      = (const float*)d_in[21];
  const float* proj_out_w  = (const float*)d_in[22];
  const float* proj_out_b  = (const float*)d_in[23];
  float* out = (float*)d_out;
  (void)in_sizes; (void)n_in; (void)out_size; (void)ws_size;

  float* ws = (float*)d_ws;
  size_t off = 0;
  float* h    = ws + off; off += (size_t)M_ * C_;
  float* ce   = ws + off; off += (size_t)M_ * C_;
  float* a    = ws + off; off += (size_t)M_ * C_;
  float* qkv  = ws + off; off += (size_t)M_ * QKVN;
  float* o    = ws + off; off += (size_t)M_ * C_;
  float* g    = ws + off; off += (size_t)M_ * FF;
  float* dist = ws + off; off += (size_t)B_ * L_ * L_;
  float* ab   = ws + off; off += 32;
  float* logits = ws + off; off += (size_t)B_ * H_ * L_ * L_;
  unsigned short* attn = (unsigned short*)(ws + off); off += (size_t)B_ * H_ * L_ * L_ / 2;
  float* y    = ws + off; off += (size_t)M_ * 8;

  h0_kernel<<<(M_ * C_) / 256, 256, 0, stream>>>(x, proj_in_w, proj_in_b, pos_embed, h);
  gemm_bf16<<<dim3(C_ / 128, M_ / 128), 256, 0, stream>>>(
      cond, proj_cond_w, proj_cond_b, nullptr, ce, M_, C_, C_, 0);
  dist_kernel<<<(B_ * L_ * L_) / 256, 256, 0, stream>>>(x, dist);
  ab_kernel<<<1, 32, 0, stream>>>(pair_w, pair_b, pairhead_w, ab);

  for (int l = 0; l < NL_; ++l) {
    add_ln_kernel<<<M_ / 8, 256, 0, stream>>>(h, ce, ln1_s + l * C_, ln1_b + l * C_, a, 1);
    gemm_bf16<<<dim3(QKVN / 128, M_ / 128), 256, 0, stream>>>(
        a, qkv_w + (size_t)l * C_ * QKVN, qkv_b + l * QKVN, nullptr, qkv, M_, QKVN, C_, 0);
    att_logits<<<64, 256, 0, stream>>>(qkv, dist, ab, l, logits);
    att_softmax<<<B_ * H_ * L_ / 8, 256, 0, stream>>>(logits, attn);
    att_av<<<64, 256, 0, stream>>>(attn, qkv, o);
    gemm_bf16<<<dim3(C_ / 128, M_ / 128), 256, 0, stream>>>(
        o, out_w + (size_t)l * C_ * C_, out_b + l * C_, h, h, M_, C_, C_, 0);
    add_ln_kernel<<<M_ / 8, 256, 0, stream>>>(h, ce, ln2_s + l * C_, ln2_b + l * C_, a, 0);
    gemm_bf16<<<dim3(FF / 128, M_ / 128), 256, 0, stream>>>(
        a, mlp1_w + (size_t)l * C_ * FF, mlp1_b + l * FF, nullptr, g, M_, FF, C_, 1);
    gemm_bf16<<<dim3(C_ / 128, M_ / 128), 256, 0, stream>>>(
        g, mlp2_w + (size_t)l * FF * C_, mlp2_b + l * C_, h, h, M_, C_, FF, 0);
  }

  proj_out_kernel<<<(M_ * 6 + 255) / 256, 256, 0, stream>>>(h, proj_out_w, proj_out_b, y);
  coupling_kernel<<<(B_ * L_ * 3 + 255) / 256, 256, 0, stream>>>(x, y, out);
  logdet_kernel<<<B_, 256, 0, stream>>>(y, out + B_ * L_ * 3);
}
